// FoundationLorentzParT_20847771254990
// MI455X (gfx1250) — compile-verified, matrix operand order checked by reading, not run on hardware
//
#include <hip/hip_runtime.h>
#include <math.h>

// ---------------------------------------------------------------------------
// FoundationLorentzParT forward for MI455X (gfx1250, wave32, WMMA bf16)
// B=128 N=128 C=128 H=8 D=16 L=8 EXP=4
// All GEMM B-operands are pre-transposed (N x K) so both A and B fragments
// are contiguous-pair LDS reads that merge into ds_load_b128.
// ---------------------------------------------------------------------------

typedef unsigned short u16;
typedef unsigned int   u32;
typedef float  v8f   __attribute__((ext_vector_type(8)));
typedef __bf16 v16bf __attribute__((ext_vector_type(16)));
typedef u16    u16x8 __attribute__((ext_vector_type(8)));

union AF { v16bf v; u32 u[8]; u16 s[16]; };

// CDNA5 16-bit A/B fragment K-index for VGPR pair v (0..7), lane half (0/1):
// K = 16*(v>=4) + 2*(v&3) + 8*half  (pair = {K, K+1})
// => lane's 16 values are two contiguous 8-element runs: [8h,8h+8) and [16+8h,16+8h+8)
#define KIDX(v, half) ((((v) & 4) << 2) + (((v) & 3) << 1) + ((half) << 3))

#define WMMA_BF16(Afr, Bfr, Cacc) \
  __builtin_amdgcn_wmma_f32_16x16x32_bf16(false, (Afr), false, (Bfr), (short)0, (Cacc), false, false)

__device__ __forceinline__ u16 f2b(float f) {
  u32 u = __float_as_uint(f);
  u32 r = u + 0x7FFFu + ((u >> 16) & 1u);   // round-to-nearest-even
  return (u16)(r >> 16);
}
__device__ __forceinline__ float b2f(u16 s) { return __uint_as_float(((u32)s) << 16); }
__device__ __forceinline__ v8f vzero8() { v8f v; for (int i = 0; i < 8; ++i) v[i] = 0.f; return v; }

// ---------------------------------------------------------------------------
// conversion kernels
// ---------------------------------------------------------------------------
__global__ void f2bf_t_kernel(const float* __restrict__ s, u16* __restrict__ d, int K, int N) {
  // src (K x N) f32 row-major -> dst (N x K) bf16 row-major; blockIdx.y = batch (layer)
  size_t off = (size_t)blockIdx.y * K * N;
  int idx = blockIdx.x * 256 + threadIdx.x;
  if (idx >= K * N) return;
  int n = idx / K, k = idx - n * K;
  d[off + idx] = f2b(s[off + (size_t)k * N + n]);
}
__global__ void padx_kernel(const float* __restrict__ x, u16* __restrict__ xp) {
  int idx = blockIdx.x * 256 + threadIdx.x;            // 16384*32
  if (idx >= 16384 * 32) return;
  int r = idx >> 5, c = idx & 31;
  xp[idx] = (c < 16) ? f2b(x[r * 16 + c]) : (u16)0;
}
__global__ void padwt_kernel(const float* __restrict__ w, u16* __restrict__ wp) {
  // proj_W (16x128) f32 -> transposed + K-padded (128 x 32) bf16
  int idx = blockIdx.x * 256 + threadIdx.x;            // 128*32
  if (idx >= 128 * 32) return;
  int n = idx >> 5, k = idx & 31;
  wp[idx] = (k < 16) ? f2b(w[k * 128 + n]) : (u16)0;
}

// ---------------------------------------------------------------------------
// Generic bf16 WMMA GEMM: C = act(A(MxK) * Bt(NxK)^T + bias) [+residual]
// tile 128x128, 256 threads = 8 waves (4 M-waves x 2 N-waves), K-step 32
// ---------------------------------------------------------------------------
#define GF_BIAS 1
#define GF_RES  2   // Cf32 += result (residual accumulate)
#define GF_STF  4   // Cf32  = result
#define GF_STB  8   // Cbf16 = bf16(result)
#define GF_GELU 16

__global__ __launch_bounds__(256) void gemm_bf16_kernel(
    const u16* __restrict__ A, int lda,
    const u16* __restrict__ Bt, int ldb,     // Bt is N x K (pre-transposed), ldb = K
    const float* __restrict__ bias,
    float* Cf, u16* Cb, int M, int N, int K, int flags) {
  __shared__ u16 As[128][40];
  __shared__ u16 Bs[128][40];
  const int tid = threadIdx.x;
  const int wave = tid >> 5, lane = tid & 31;
  const int half = lane >> 4, lm = lane & 15;
  const int m0 = blockIdx.y * 128, n0 = blockIdx.x * 128;
  const int mw = (wave >> 1) * 32;
  const int nw = (wave & 1) * 64;

  v8f acc[2][4];
  for (int i = 0; i < 2; ++i) for (int j = 0; j < 4; ++j) acc[i][j] = vzero8();

  for (int kt = 0; kt < K; kt += 32) {
    __syncthreads();
    for (int it = 0; it < 2; ++it) {                    // A tile 128x32
      int idx = tid + it * 256;
      int r = idx >> 2, c = (idx & 3) << 3;
      *(u16x8*)&As[r][c] = *(const u16x8*)&A[(size_t)(m0 + r) * lda + kt + c];
    }
    for (int it = 0; it < 2; ++it) {                    // Bt tile 128x32
      int idx = tid + it * 256;
      int r = idx >> 2, c = (idx & 3) << 3;
      *(u16x8*)&Bs[r][c] = *(const u16x8*)&Bt[(size_t)(n0 + r) * ldb + kt + c];
    }
    __syncthreads();

    AF af[2], bf[4];
    for (int mb = 0; mb < 2; ++mb) {
      int row = mw + mb * 16 + lm;
      for (int v = 0; v < 8; ++v) af[mb].u[v] = *(const u32*)&As[row][KIDX(v, half)];
    }
    for (int nb = 0; nb < 4; ++nb) {
      int n = nw + nb * 16 + lm;
      for (int v = 0; v < 8; ++v) bf[nb].u[v] = *(const u32*)&Bs[n][KIDX(v, half)];
    }
    for (int nb = 0; nb < 4; ++nb)
      for (int mb = 0; mb < 2; ++mb)
        acc[mb][nb] = WMMA_BF16(af[mb].v, bf[nb].v, acc[mb][nb]);
  }

  for (int mb = 0; mb < 2; ++mb)
    for (int nb = 0; nb < 4; ++nb) {
      int n = n0 + nw + nb * 16 + lm;
      float bv = (flags & GF_BIAS) ? bias[n] : 0.f;
      for (int e = 0; e < 8; ++e) {
        int m = m0 + mw + mb * 16 + e + 8 * half;
        float v = acc[mb][nb][e] + bv;
        if (flags & GF_GELU) v = 0.5f * v * (1.f + erff(v * 0.70710678118654752f));
        size_t off = (size_t)m * N + n;
        if (flags & GF_RES) Cf[off] += v;
        else if (flags & GF_STF) Cf[off] = v;
        if (flags & GF_STB) Cb[off] = f2b(v);
      }
    }
}

// ---------------------------------------------------------------------------
// LayerNorm: one wave per 128-wide row; bf16 and/or f32 outputs
// ---------------------------------------------------------------------------
__global__ __launch_bounds__(256) void ln_kernel(
    const float* __restrict__ X, const float* __restrict__ g, const float* __restrict__ bta,
    u16* __restrict__ Yb, float* __restrict__ Yf) {
  int row = blockIdx.x * 8 + (threadIdx.x >> 5);
  int lane = threadIdx.x & 31;
  const float* xr = X + (size_t)row * 128 + lane * 4;
  float4 t = *(const float4*)xr;
  float xv[4] = {t.x, t.y, t.z, t.w};
  float s = xv[0] + xv[1] + xv[2] + xv[3];
  for (int m = 1; m < 32; m <<= 1) s += __shfl_xor(s, m, 32);
  float mu = s * (1.f / 128.f);
  float vs = 0.f;
  for (int c = 0; c < 4; ++c) { float d = xv[c] - mu; vs += d * d; }
  for (int m = 1; m < 32; m <<= 1) vs += __shfl_xor(vs, m, 32);
  float inv = rsqrtf(vs * (1.f / 128.f) + 1e-5f);
  int base = lane * 4;
  float o[4];
  for (int c = 0; c < 4; ++c) o[c] = (xv[c] - mu) * inv * g[base + c] + bta[base + c];
  if (Yb) {
    uint2 pp;
    pp.x = (u32)f2b(o[0]) | ((u32)f2b(o[1]) << 16);
    pp.y = (u32)f2b(o[2]) | ((u32)f2b(o[3]) << 16);
    *(uint2*)&Yb[(size_t)row * 128 + base] = pp;
  }
  if (Yf) {
    float4 ff; ff.x = o[0]; ff.y = o[1]; ff.z = o[2]; ff.w = o[3];
    *(float4*)&Yf[(size_t)row * 128 + base] = ff;
  }
}

// ---------------------------------------------------------------------------
// Pair MLP: per (b,i) block; rows = j (128). 4->64->64->64->8, ReLU each.
// Layers 2,3 via LDS-resident 128x64x64 WMMA GEMMs (weights pre-transposed).
// ---------------------------------------------------------------------------
__device__ __forceinline__ void lds_gemm64(
    u16 (*Hin)[72], u16 (*WsT)[72], const float* __restrict__ bias,
    u16 (*Hout)[72], int wave, int lane) {
  const int half = lane >> 4, lm = lane & 15;
  v8f acc[4];
  for (int j = 0; j < 4; ++j) acc[j] = vzero8();
  for (int kk = 0; kk < 2; ++kk) {
    AF a, bf[4];
    int row = wave * 16 + lm;
    for (int v = 0; v < 8; ++v)
      a.u[v] = *(const u32*)&Hin[row][kk * 32 + KIDX(v, half)];
    for (int nb = 0; nb < 4; ++nb) {
      int n = nb * 16 + lm;
      for (int v = 0; v < 8; ++v)
        bf[nb].u[v] = *(const u32*)&WsT[n][kk * 32 + KIDX(v, half)];
    }
    for (int nb = 0; nb < 4; ++nb)
      acc[nb] = WMMA_BF16(a.v, bf[nb].v, acc[nb]);
  }
  for (int nb = 0; nb < 4; ++nb) {
    int n = nb * 16 + lm;
    float bv = bias[n];
    for (int e = 0; e < 8; ++e) {
      int m = wave * 16 + e + 8 * half;
      Hout[m][n] = f2b(fmaxf(acc[nb][e] + bv, 0.f));
    }
  }
}

__global__ __launch_bounds__(256) void pair_kernel(
    const float* __restrict__ U,
    const float* __restrict__ W1, const float* __restrict__ b1,
    const u16* __restrict__ W2t, const float* __restrict__ b2,
    const u16* __restrict__ W3t, const float* __restrict__ b3,
    const float* __restrict__ W4, const float* __restrict__ b4,
    float* __restrict__ biasg) {
  __shared__ u16 W2s[64][72];
  __shared__ u16 W3s[64][72];
  __shared__ u16 H0[128][72];
  __shared__ u16 H1[128][72];
  const int tid = threadIdx.x, wave = tid >> 5, lane = tid & 31;
  const int b = blockIdx.x >> 7, i = blockIdx.x & 127;

  for (int idx = tid; idx < 512; idx += 256) {          // 4096 elems, 8-wide chunks
    int r = idx >> 3, c = (idx & 7) << 3;
    *(u16x8*)&W2s[r][c] = *(const u16x8*)&W2t[r * 64 + c];
    *(u16x8*)&W3s[r][c] = *(const u16x8*)&W3t[r * 64 + c];
  }
  {   // layer 1: 4 -> 64, scalar
    int j = tid >> 1;
    int cbase = (tid & 1) * 32;
    const float* up = &U[(((size_t)(b * 128 + i)) * 128 + j) * 4];
    float u0 = up[0], u1 = up[1], u2 = up[2], u3 = up[3];
    for (int cc = 0; cc < 32; ++cc) {
      int c = cbase + cc;
      float a = b1[c] + u0 * W1[c] + u1 * W1[64 + c] + u2 * W1[128 + c] + u3 * W1[192 + c];
      H0[j][c] = f2b(fmaxf(a, 0.f));
    }
  }
  __syncthreads();
  lds_gemm64(H0, W2s, b2, H1, wave, lane);
  __syncthreads();
  lds_gemm64(H1, W3s, b3, H0, wave, lane);
  __syncthreads();
  // layer 4: 64 -> 8 (scalar) + transposed store -> attn_bias[b][h][i][j]
  for (int idx = tid; idx < 1024; idx += 256) {
    int hh = idx >> 7, j = idx & 127;
    float a = b4[hh];
    for (int c = 0; c < 64; ++c) a += b2f(H0[j][c]) * W4[c * 8 + hh];
    a = fmaxf(a, 0.f);
    biasg[(((size_t)(b * 8 + hh)) * 128 + i) * 128 + j] = a;
  }
}

// ---------------------------------------------------------------------------
// Fused attention per (b,h): S = QK^T*0.25 + bias + mask; softmax; O = P V
// K rows staged row-major (Bt of S-GEMM is just k[b,j,h,:]); V staged transposed.
// ---------------------------------------------------------------------------
__global__ __launch_bounds__(256) void attn_kernel(
    const u16* __restrict__ qbf, const u16* __restrict__ kbf, const u16* __restrict__ vbf,
    const float* __restrict__ biasg, const unsigned char* __restrict__ mask,
    u16* __restrict__ obf) {
  __shared__ u16 Qs[128][24];
  __shared__ u16 Ks[128][24];
  __shared__ u16 Vt[16][136];
  __shared__ u16 Ps[8][16][136];
  const int tid = threadIdx.x, wave = tid >> 5, lane = tid & 31;
  const int half = lane >> 4, lm = lane & 15;
  const int b = blockIdx.x >> 3, hh = blockIdx.x & 7;

  {   // stage Q, K row-major; V transposed (Vt[d][j])
    int i = tid >> 1, c = (tid & 1) << 3;
    size_t base = ((size_t)(b * 128 + i)) * 128 + hh * 16 + c;
    *(u16x8*)&Qs[i][c] = *(const u16x8*)&qbf[base];
    *(u16x8*)&Ks[i][c] = *(const u16x8*)&kbf[base];
    u16x8 vv = *(const u16x8*)&vbf[base];
    for (int dd = 0; dd < 8; ++dd) Vt[c + dd][i] = vv[dd];
  }
  __syncthreads();

  // S: wave owns rows [16*wave, 16*wave+16); K-dim padded 16 -> 32 with zeros
  v8f acc[8];
  for (int j = 0; j < 8; ++j) acc[j] = vzero8();
  AF aq;
  int row = wave * 16 + lm;
  for (int v = 0; v < 8; ++v)
    aq.u[v] = (v < 4) ? *(const u32*)&Qs[row][KIDX(v, half)] : 0u;
  for (int j8 = 0; j8 < 8; ++j8) {
    AF bk;
    int n = j8 * 16 + lm;
    for (int v = 0; v < 8; ++v)
      bk.u[v] = (v < 4) ? *(const u32*)&Ks[n][KIDX(v, half)] : 0u;
    acc[j8] = WMMA_BF16(aq.v, bk.v, acc[j8]);
  }

  // bias + mask + in-wave softmax (row lives in one 16-lane half)
  float madd[8];
  for (int j8 = 0; j8 < 8; ++j8)
    madd[j8] = mask[b * 128 + j8 * 16 + lm] ? 0.f : -__builtin_inff();
  const float* brow = biasg + ((size_t)(b * 8 + hh)) * 128 * 128;
  for (int e = 0; e < 8; ++e) {
    int i = wave * 16 + e + 8 * half;
    const float* br = brow + (size_t)i * 128;
    float vv[8], rm = -__builtin_inff();
    for (int j8 = 0; j8 < 8; ++j8) {
      float v = acc[j8][e] * 0.25f + br[j8 * 16 + lm] + madd[j8];
      vv[j8] = v; rm = fmaxf(rm, v);
    }
    for (int m2 = 1; m2 < 16; m2 <<= 1) rm = fmaxf(rm, __shfl_xor(rm, m2, 32));
    float sum = 0.f;
    for (int j8 = 0; j8 < 8; ++j8) {
      float p = (rm <= -1e30f) ? 0.f : expf(vv[j8] - rm);
      vv[j8] = p; sum += p;
    }
    for (int m2 = 1; m2 < 16; m2 <<= 1) sum += __shfl_xor(sum, m2, 32);
    float inv = (sum > 0.f) ? 1.f / sum : 0.f;
    for (int j8 = 0; j8 < 8; ++j8)
      Ps[wave][e + 8 * half][j8 * 16 + lm] = f2b(vv[j8] * inv);
  }
  __syncthreads();

  // O = P(16x128) @ V(128x16): 4 K-steps of 32; B-frag rows from Vt are contiguous
  v8f oacc = vzero8();
  for (int kk = 0; kk < 4; ++kk) {
    AF ap, bv;
    for (int v = 0; v < 8; ++v)
      ap.u[v] = *(const u32*)&Ps[wave][lm][kk * 32 + KIDX(v, half)];
    for (int v = 0; v < 8; ++v)
      bv.u[v] = *(const u32*)&Vt[lm][kk * 32 + KIDX(v, half)];
    oacc = WMMA_BF16(ap.v, bv.v, oacc);
  }
  for (int e = 0; e < 8; ++e) {
    int i = wave * 16 + e + 8 * half;
    obf[((size_t)(b * 128 + i)) * 128 + hh * 16 + lm] = f2b(oacc[e]);
  }
}

// ---------------------------------------------------------------------------
// host-side launch
// ---------------------------------------------------------------------------
extern "C" void kernel_launch(void* const* d_in, const int* in_sizes, int n_in,
                              void* d_out, int out_size, void* d_ws, size_t ws_size,
                              hipStream_t stream) {
  // top-level insertion order; params flattened jax-tree style (sorted dict keys)
  enum { I_X = 0, I_MASK = 1, I_U = 2,
         I_WF1 = 3, I_WF2 = 4, I_WK = 5, I_WO = 6, I_WQ = 7, I_WV = 8,
         I_BF1 = 9, I_BF2 = 10, I_BK = 11, I_BO = 12, I_BQ = 13, I_BV = 14,
         I_LN1B = 15, I_LN1G = 16, I_LN2B = 17, I_LN2G = 18,
         I_LNOB = 19, I_LNOG = 20,
         I_PW1 = 21, I_PB1 = 22, I_PW2 = 23, I_PB2 = 24,
         I_PW3 = 25, I_PB3 = 26, I_PW4 = 27, I_PB4 = 28,
         I_PRW = 29, I_PRB = 30 };
  const int M = 16384, C = 128, L = 8;

  char* ws = (char*)d_ws;
  size_t o = 0;
  auto alloc = [&](size_t bytes) { char* p = ws + o; o += (bytes + 255) & ~(size_t)255; return p; };
  float* h     = (float*)alloc((size_t)M * C * 4);
  u16*   ybf   = (u16*)  alloc((size_t)M * C * 2);
  u16*   qbf   = (u16*)  alloc((size_t)M * C * 2);
  u16*   kbf   = (u16*)  alloc((size_t)M * C * 2);
  u16*   vbf   = (u16*)  alloc((size_t)M * C * 2);
  u16*   obf   = (u16*)  alloc((size_t)M * C * 2);
  u16*   fbf   = (u16*)  alloc((size_t)M * 512 * 2);
  u16*   xpad  = (u16*)  alloc((size_t)M * 32 * 2);
  u16*   wproj = (u16*)  alloc(128 * 32 * 2);          // transposed, K-padded
  u16*   wq    = (u16*)  alloc((size_t)L * 16384 * 2); // all transposed (N x K)
  u16*   wk    = (u16*)  alloc((size_t)L * 16384 * 2);
  u16*   wv    = (u16*)  alloc((size_t)L * 16384 * 2);
  u16*   wo    = (u16*)  alloc((size_t)L * 16384 * 2);
  u16*   wf1   = (u16*)  alloc((size_t)L * 65536 * 2);
  u16*   wf2   = (u16*)  alloc((size_t)L * 65536 * 2);
  u16*   wp2   = (u16*)  alloc(4096 * 2);
  u16*   wp3   = (u16*)  alloc(4096 * 2);
  float* abias = (float*)alloc((size_t)128 * 8 * 128 * 128 * 4);

  auto cvtT = [&](int idx, u16* dst, int K, int N, int batch) {
    dim3 g((K * N + 255) / 256, batch);
    f2bf_t_kernel<<<g, 256, 0, stream>>>((const float*)d_in[idx], dst, K, N);
  };
  cvtT(I_WQ, wq, 128, 128, L);   cvtT(I_WK, wk, 128, 128, L);
  cvtT(I_WV, wv, 128, 128, L);   cvtT(I_WO, wo, 128, 128, L);
  cvtT(I_WF1, wf1, 128, 512, L); cvtT(I_WF2, wf2, 512, 128, L);
  cvtT(I_PW2, wp2, 64, 64, 1);   cvtT(I_PW3, wp3, 64, 64, 1);
  padx_kernel<<<(M * 32 + 255) / 256, 256, 0, stream>>>((const float*)d_in[I_X], xpad);
  padwt_kernel<<<(128 * 32 + 255) / 256, 256, 0, stream>>>((const float*)d_in[I_PRW], wproj);

  pair_kernel<<<16384, 256, 0, stream>>>(
      (const float*)d_in[I_U],
      (const float*)d_in[I_PW1], (const float*)d_in[I_PB1],
      wp2, (const float*)d_in[I_PB2],
      wp3, (const float*)d_in[I_PB3],
      (const float*)d_in[I_PW4], (const float*)d_in[I_PB4], abias);

  // h = x @ proj_W + proj_b  (K padded 16 -> 32)
  gemm_bf16_kernel<<<dim3(1, 128), 256, 0, stream>>>(
      xpad, 32, wproj, 32, (const float*)d_in[I_PRB],
      h, nullptr, M, 128, 32, GF_BIAS | GF_STF);

  const unsigned char* mask = (const unsigned char*)d_in[I_MASK];
  for (int l = 0; l < L; ++l) {
    const float* ln1g = (const float*)d_in[I_LN1G] + l * C;
    const float* ln1b = (const float*)d_in[I_LN1B] + l * C;
    const float* ln2g = (const float*)d_in[I_LN2G] + l * C;
    const float* ln2b = (const float*)d_in[I_LN2B] + l * C;

    ln_kernel<<<2048, 256, 0, stream>>>(h, ln1g, ln1b, ybf, nullptr);

    gemm_bf16_kernel<<<dim3(1, 128), 256, 0, stream>>>(
        ybf, 128, wq + (size_t)l * 16384, 128, (const float*)d_in[I_BQ] + l * C,
        nullptr, qbf, M, 128, 128, GF_BIAS | GF_STB);
    gemm_bf16_kernel<<<dim3(1, 128), 256, 0, stream>>>(
        ybf, 128, wk + (size_t)l * 16384, 128, (const float*)d_in[I_BK] + l * C,
        nullptr, kbf, M, 128, 128, GF_BIAS | GF_STB);
    gemm_bf16_kernel<<<dim3(1, 128), 256, 0, stream>>>(
        ybf, 128, wv + (size_t)l * 16384, 128, (const float*)d_in[I_BV] + l * C,
        nullptr, vbf, M, 128, 128, GF_BIAS | GF_STB);

    attn_kernel<<<1024, 256, 0, stream>>>(qbf, kbf, vbf, abias, mask, obf);

    gemm_bf16_kernel<<<dim3(1, 128), 256, 0, stream>>>(
        obf, 128, wo + (size_t)l * 16384, 128, (const float*)d_in[I_BO] + l * C,
        h, nullptr, M, 128, 128, GF_BIAS | GF_RES);

    ln_kernel<<<2048, 256, 0, stream>>>(h, ln2g, ln2b, ybf, nullptr);

    gemm_bf16_kernel<<<dim3(4, 128), 256, 0, stream>>>(
        ybf, 128, wf1 + (size_t)l * 65536, 128, (const float*)d_in[I_BF1] + l * 512,
        nullptr, fbf, M, 512, 128, GF_BIAS | GF_STB | GF_GELU);
    gemm_bf16_kernel<<<dim3(1, 128), 256, 0, stream>>>(
        fbf, 512, wf2 + (size_t)l * 65536, 512, (const float*)d_in[I_BF2] + l * C,
        h, nullptr, M, 128, 512, GF_BIAS | GF_RES);
  }

  ln_kernel<<<2048, 256, 0, stream>>>(
      h, (const float*)d_in[I_LNOG], (const float*)d_in[I_LNOB], nullptr, (float*)d_out);

  (void)in_sizes; (void)n_in; (void)out_size; (void)ws_size;
}
